// AUGRU_57999238365393
// MI455X (gfx1250) — matrix operand order repaired and checked by reading, not compiled, not run on hardware
//
#include <hip/hip_runtime.h>

// Fused AUGRU for MI455X (gfx1250): weights-stationary in LDS (f16, fragment-
// friendly transposed layout), 32 batch rows per workgroup, 200 sequential
// steps of v_wmma_f32_16x16x32_f16 GEMMs with f32 hidden state in LDS.
// r*h is fused into the phase-1 epilogue; x_{t+1} is prefetched during phase 0.

typedef _Float16 v4h  __attribute__((ext_vector_type(4)));
typedef _Float16 v8h  __attribute__((ext_vector_type(8)));
typedef _Float16 v16h __attribute__((ext_vector_type(16)));
typedef float    v8f  __attribute__((ext_vector_type(8)));

#define BT 32     // batch rows per workgroup
#define DH 128    // D == H
#define WS 264    // weight column stride in halves (256 + 8 pad -> breaks 512B bank stride)
#define CS 264    // comb row stride (halves)
#define ZS 136    // z row stride (halves)
#define HS 132    // f32 h-state row stride (floats)

__device__ __forceinline__ v16h mk16(v8h lo, v8h hi) {
  return __builtin_shufflevector(lo, hi, 0,1,2,3,4,5,6,7,8,9,10,11,12,13,14,15);
}
__device__ __forceinline__ float sigm_(float x) { return 1.0f / (1.0f + __expf(-x)); }
__device__ __forceinline__ float tanh_(float x) { return 1.0f - 2.0f / (__expf(2.0f * x) + 1.0f); }

__global__ __launch_bounds__(256, 1)
void augru_fused(const float* __restrict__ x,    // [B,T,D]
                 const float* __restrict__ att,  // [B,T]
                 const float* __restrict__ h0,   // [B,H]
                 const float* __restrict__ Wz, const float* __restrict__ bz,
                 const float* __restrict__ Wr, const float* __restrict__ br,
                 const float* __restrict__ Wh, const float* __restrict__ bh,
                 float* __restrict__ out, int B, int T)
{
  __shared__ _Float16 Wlds[384 * WS];   // cols 0-127:Wz 128-255:Wr 256-383:Wh, [col][K]
  __shared__ _Float16 comb[BT * CS];    // [row][256] = [x_t | h (or r*h)]
  __shared__ _Float16 zbuf[BT * ZS];    // [row][128] = z post-sigmoid
  __shared__ float    h32 [BT * HS];    // f32 hidden state
  __shared__ float    bzr [256];
  __shared__ float    bhs [128];
  __shared__ float    aBuf[BT];

  const int tid  = threadIdx.x;
  const int lane = tid & 31;
  const int w    = tid >> 5;     // wave id 0..7
  const int ln   = lane & 15;
  const int lh   = lane >> 4;
  const int b0   = blockIdx.x * BT;

  // ---- one-time setup: weights (transposed), biases, h0 ----
  for (int e = tid; e < 3 * DH * 256; e += 256) {
    int g   = e >> 15;           // gate: 0=z 1=r 2=h
    int rem = e & 32767;
    int k   = rem >> 7;          // K row 0..255
    int col = rem & 127;         // output col (coalesced reads)
    const float* Wg = (g == 0) ? Wz : (g == 1) ? Wr : Wh;
    Wlds[(g * DH + col) * WS + k] = (_Float16)Wg[k * DH + col];
  }
  if (tid < 128) { bzr[tid] = bz[tid]; bzr[tid + 128] = br[tid]; bhs[tid] = bh[tid]; }
  for (int e = tid; e < BT * DH; e += 256) {
    int row = e >> 7, col = e & 127;
    float v = h0[(size_t)(b0 + row) * DH + col];
    h32[row * HS + col] = v;
    comb[row * CS + DH + col] = (_Float16)v;
  }
  __syncthreads();

  for (int t = 0; t < T; ++t) {
    // ---- phase 0: stream x_t tile -> comb[:,0:128] (f16); a_t -> aBuf;
    //      prefetch x_{t+1} behind this step's compute ----
    {
      const float4* x4 = (const float4*)x;
      #pragma unroll
      for (int i = 0; i < 4; ++i) {
        int e   = tid + i * 256;       // 0..1023 over 32 rows x 32 float4
        int row = e >> 5;
        int c4  = e & 31;
        float4 v = x4[((size_t)(b0 + row) * T + t) * (DH / 4) + c4];
        v4h hv = { (_Float16)v.x, (_Float16)v.y, (_Float16)v.z, (_Float16)v.w };
        *(v4h*)&comb[row * CS + c4 * 4] = hv;
        if ((c4 & 3) == 0 && t + 1 < T)   // one prefetch per 64B line
          __builtin_prefetch(&x4[((size_t)(b0 + row) * T + t + 1) * (DH / 4) + c4], 0, 3);
      }
      if (tid < BT) aBuf[tid] = att[(size_t)(b0 + tid) * T + t];
    }
    __syncthreads();

    // ---- phase 1: [32,256] x [256,256] -> z|r ----
    {
      const int mt   = w & 1;
      const int ntb  = (w >> 1) * 4;    // all 4 tiles of a wave are gate-uniform
      const int arow = mt * 16 + ln;
      v8f acc[4] = {};
      #pragma unroll
      for (int k = 0; k < 8; ++k) {
        const _Float16* pa = &comb[arow * CS + k * 32 + lh * 8];
        v16h Ak = mk16(*(const v8h*)pa, *(const v8h*)(pa + 16));
        #pragma unroll
        for (int j = 0; j < 4; ++j) {
          const int n = (ntb + j) * 16 + ln;
          const _Float16* pb = &Wlds[n * WS + k * 32 + lh * 8];
          v16h Bf = mk16(*(const v8h*)pb, *(const v8h*)(pb + 16));
          acc[j] = __builtin_amdgcn_wmma_f32_16x16x32_f16(
              false, Ak, false, Bf, (short)0, acc[j], false, false);
        }
      }
      __syncthreads();   // all waves done reading comb before we overwrite h-cols

      // epilogue: z -> zbuf ; r -> comb_h = sigmoid(r)*h (fused, no phase 2)
      #pragma unroll
      for (int j = 0; j < 4; ++j) {
        const int col = (ntb + j) * 16 + ln;
        const float bb = bzr[col];
        if (col < 128) {                       // wave-uniform branch
          #pragma unroll
          for (int i = 0; i < 8; ++i) {
            int row = mt * 16 + lh * 8 + i;
            zbuf[row * ZS + col] = (_Float16)sigm_(acc[j][i] + bb);
          }
        } else {
          #pragma unroll
          for (int i = 0; i < 8; ++i) {
            int row = mt * 16 + lh * 8 + i;
            float rv = sigm_(acc[j][i] + bb);
            comb[row * CS + col] = (_Float16)(rv * h32[row * HS + col - 128]);
          }
        }
      }
    }
    __syncthreads();

    // ---- phase 3a: h_tilde GEMM [32,256] x [256,128] ----
    v8f hacc[2] = {};
    {
      const int mt   = w & 1;
      const int ntb  = (w >> 1) * 2;
      const int arow = mt * 16 + ln;
      #pragma unroll
      for (int k = 0; k < 8; ++k) {
        const _Float16* pa = &comb[arow * CS + k * 32 + lh * 8];
        v16h Ak = mk16(*(const v8h*)pa, *(const v8h*)(pa + 16));
        #pragma unroll
        for (int j = 0; j < 2; ++j) {
          const int n = 256 + (ntb + j) * 16 + ln;   // Wh columns
          const _Float16* pb = &Wlds[n * WS + k * 32 + lh * 8];
          v16h Bf = mk16(*(const v8h*)pb, *(const v8h*)(pb + 16));
          hacc[j] = __builtin_amdgcn_wmma_f32_16x16x32_f16(
              false, Ak, false, Bf, (short)0, hacc[j], false, false);
        }
      }
    }
    __syncthreads();

    // ---- phase 3b: h_new = (1 - a*z)*h + a*z*tanh(.) ; update state ----
    {
      const int mt  = w & 1;
      const int ntb = (w >> 1) * 2;
      #pragma unroll
      for (int j = 0; j < 2; ++j) {
        const int col = (ntb + j) * 16 + ln;
        const float bb = bhs[col];
        #pragma unroll
        for (int i = 0; i < 8; ++i) {
          int row = mt * 16 + lh * 8 + i;
          float ht   = tanh_(hacc[j][i] + bb);
          float zp   = aBuf[row] * (float)zbuf[row * ZS + col];
          float hold = h32[row * HS + col];
          float hn   = fmaf(zp, ht - hold, hold);
          h32[row * HS + col] = hn;
          comb[row * CS + DH + col] = (_Float16)hn;
        }
      }
    }
    __syncthreads();
  }

  // ---- epilogue: write final h (f32) ----
  #pragma unroll
  for (int i = 0; i < 16; ++i) {
    int e = tid + i * 256;
    int row = e >> 7, col = e & 127;
    out[(size_t)(b0 + row) * DH + col] = h32[row * HS + col];
  }
}

extern "C" void kernel_launch(void* const* d_in, const int* in_sizes, int n_in,
                              void* d_out, int out_size, void* d_ws, size_t ws_size,
                              hipStream_t stream) {
  const float* x   = (const float*)d_in[0];
  const float* att = (const float*)d_in[1];
  const float* h0  = (const float*)d_in[2];
  const float* Wz  = (const float*)d_in[3];
  const float* bz  = (const float*)d_in[4];
  const float* Wr  = (const float*)d_in[5];
  const float* br  = (const float*)d_in[6];
  const float* Wh  = (const float*)d_in[7];
  const float* bh  = (const float*)d_in[8];
  float* out = (float*)d_out;

  const int B = in_sizes[2] / DH;        // h0 is [B,H]
  const int T = in_sizes[1] / B;         // attention is [B,T]

  dim3 grid(B / BT), block(256);
  augru_fused<<<grid, block, 0, stream>>>(x, att, h0, Wz, bz, Wr, br, Wh, bh, out, B, T);
}
